// Attention1DOnnx_48241072669074
// MI455X (gfx1250) — compile-verified, matrix operand order checked by reading, not run on hardware
//
#include <hip/hip_runtime.h>
#include <hip/hip_bf16.h>

// ---------------------------------------------------------------------------
// Windowed 1D attention (K=17, R=8), B=2, C=128, H=64, W=256, gfx1250.
// Fused design: per-(b,h)-row workgroup computes Q/K via WMMA straight into
// LDS (q/kf never hit HBM), then does softmax-attention from LDS.
// B-operand row staged into LDS via global_load_async_to_lds_b128 (ASYNCcnt).
// ---------------------------------------------------------------------------

#define B_   2
#define C_   128
#define H_   64
#define W_   256
#define HW_  (H_ * W_)
#define N_   (B_ * H_ * W_)      // 32768 pixels
#define KWIN 17
#define RAD  8

#define QSTR 136                 // f16 row stride for sq/skf/xB (272 B = 17*16)
#define VSTR 129                 // f32 row stride for value row (bank-conflict-free)

typedef _Float16 v4h  __attribute__((ext_vector_type(4)));
typedef _Float16 v8h  __attribute__((ext_vector_type(8)));
typedef _Float16 v16h __attribute__((ext_vector_type(16)));
typedef float    v8f  __attribute__((ext_vector_type(8)));
typedef int      v4i  __attribute__((ext_vector_type(4)));

typedef __attribute__((address_space(1))) v4i* gptr_b128;   // global int4*
typedef __attribute__((address_space(3))) v4i* lptr_b128;   // LDS int4*

#if defined(__has_builtin)
#if __has_builtin(__builtin_amdgcn_global_load_async_to_lds_b128) && \
    __has_builtin(__builtin_amdgcn_s_wait_asynccnt)
#define HAVE_ASYNC_LDS 1
#endif
#endif

// ---------------------------------------------------------------------------
// Kernel 1: weights fp32 -> f16; fold 1/sqrt(C) into Wq and bq.
// ---------------------------------------------------------------------------
__global__ void prep_w(const float* __restrict__ Wq, const float* __restrict__ Wk,
                       const float* __restrict__ bq,
                       _Float16* __restrict__ wqh, _Float16* __restrict__ wkh,
                       float* __restrict__ bqs) {
  const float inv_s = 0.088388347648318447f;  // 1/sqrt(128)
  int i = blockIdx.x * blockDim.x + threadIdx.x;   // 64 blocks * 256 = 16384
  wqh[i] = (_Float16)(Wq[i] * inv_s);
  wkh[i] = (_Float16)(Wk[i]);
  if (i < C_) bqs[i] = bq[i] * inv_s;
}

// ---------------------------------------------------------------------------
// Kernel 2: xh = (f16)(feature + position), pixel-major [N,128] for WMMA B.
// ---------------------------------------------------------------------------
__global__ void pack_x(const float* __restrict__ feat, const float* __restrict__ pos,
                       _Float16* __restrict__ xh) {
  int n  = blockIdx.x * blockDim.x + threadIdx.x;  // 0..N_-1
  int b  = n / HW_;
  int hw = n - b * HW_;
  const float* f = feat + (size_t)b * C_ * HW_ + hw;
  const float* p = pos  + (size_t)b * C_ * HW_ + hw;
  _Float16* hr = xh + (size_t)n * C_;
#pragma unroll
  for (int c0 = 0; c0 < C_; c0 += 4) {
    float4 v;
    v.x = f[(size_t)(c0 + 0) * HW_] + p[(size_t)(c0 + 0) * HW_];
    v.y = f[(size_t)(c0 + 1) * HW_] + p[(size_t)(c0 + 1) * HW_];
    v.z = f[(size_t)(c0 + 2) * HW_] + p[(size_t)(c0 + 2) * HW_];
    v.w = f[(size_t)(c0 + 3) * HW_] + p[(size_t)(c0 + 3) * HW_];
    v4h hv = { (_Float16)v.x, (_Float16)v.y, (_Float16)v.z, (_Float16)v.w };
    *(v4h*)(hr + c0) = hv;
  }
}

// ---------------------------------------------------------------------------
// Kernel 3 (fused): per-row Q/K GEMM on WMMA + windowed softmax attention.
//
// Phase 0: async-stage the row's xh [256][128] f16 into LDS (xB, stride 136).
// Phase 1: wave m computes o-tile m for all 16 n-tiles; A operands (Wq, Wk)
//          preloaded in registers (ISA 7.12.2 16-bit A layout: lane l%16 = row,
//          VGPR0..3 = K kbase..kbase+7, VGPR4..7 = K kbase+16..23, kbase =
//          8*(l>=16); B mirrors A with n in lanes). D: (m = r+8*(l>=16),
//          n = l%16). Accumulators + bias -> f16 rows sq/skf in LDS.
// Phase 2: value row rebuilt fp32 from feature+position (coalesced) into the
//          region that previously held xB; per-pixel 17-tap scores, softmax
//          (zero-padded taps score exactly 0 -> matches reference), output.
// ---------------------------------------------------------------------------
__global__ void attn_fused(const _Float16* __restrict__ xh,
                           const _Float16* __restrict__ wqh,
                           const _Float16* __restrict__ wkh,
                           const float* __restrict__ bqs, const float* __restrict__ bk,
                           const float* __restrict__ feat, const float* __restrict__ pos,
                           float* __restrict__ out) {
  extern __shared__ char lds[];
  _Float16* sq  = (_Float16*)(lds);                      // [256][136] f16
  _Float16* skf = (_Float16*)(lds + 69632);              // [256][136] f16
  _Float16* xB  = (_Float16*)(lds + 139264);             // [256][136] f16 (phase 0/1)
  float*    sxv = (float*)(lds + 139264);                // [256][129] f32 (phase 2)

  const int bh  = blockIdx.x;        // 0..127
  const int b   = bh / H_;
  const int h   = bh - b * H_;
  const int n0  = bh * W_;           // global pixel n = n0 + w
  const int tid = threadIdx.x;       // 0..255

  // ---- Phase 0: stage xh row into LDS (4096 16B chunks, 16 per thread) ----
#if defined(HAVE_ASYNC_LDS)
  for (int i = tid; i < W_ * (C_ / 8); i += W_) {
    const int row = i >> 4;
    const int ch  = (i & 15) * 8;
    _Float16* gp = const_cast<_Float16*>(xh) + (size_t)(n0 + row) * C_ + ch;
    _Float16* lp = xB + row * QSTR + ch;
    __builtin_amdgcn_global_load_async_to_lds_b128((gptr_b128)gp, (lptr_b128)lp, 0, 0);
  }
  __builtin_amdgcn_s_wait_asynccnt(0);
#else
  for (int i = tid; i < W_ * (C_ / 8); i += W_) {
    const int row = i >> 4;
    const int ch  = (i & 15) * 8;
    *(v8h*)(xB + row * QSTR + ch) = *(const v8h*)(xh + (size_t)(n0 + row) * C_ + ch);
  }
#endif
  __syncthreads();

  // ---- Phase 1: WMMA q/kf tiles -> LDS ----
  {
    const int lane = tid & 31;
    const int wave = tid >> 5;        // o-tile 0..7
    const int lm   = lane & 15;
    const int half = lane >> 4;
    const int kb   = half * 8;
    const int mcol = wave * 16 + half * 8;

    const _Float16* wqr = wqh + (size_t)(wave * 16 + lm) * C_;
    const _Float16* wkr = wkh + (size_t)(wave * 16 + lm) * C_;

    v16h aq[4], ak[4];
#pragma unroll
    for (int kk = 0; kk < 4; ++kk) {
      const int c0 = kk * 32 + kb;
      v8h qlo = *(const v8h*)(wqr + c0);
      v8h qhi = *(const v8h*)(wqr + c0 + 16);
      aq[kk] = __builtin_shufflevector(qlo, qhi, 0,1,2,3,4,5,6,7,8,9,10,11,12,13,14,15);
      v8h klo = *(const v8h*)(wkr + c0);
      v8h khi = *(const v8h*)(wkr + c0 + 16);
      ak[kk] = __builtin_shufflevector(klo, khi, 0,1,2,3,4,5,6,7,8,9,10,11,12,13,14,15);
    }

    float bq8[8], bk8[8];
#pragma unroll
    for (int r = 0; r < 8; ++r) {
      bq8[r] = bqs[mcol + r];
      bk8[r] = bk[mcol + r];
    }

    for (int nt = 0; nt < 16; ++nt) {
      const _Float16* xr = xB + (nt * 16 + lm) * QSTR;
      v8f accq = {};
      v8f acck = {};
#pragma unroll
      for (int kk = 0; kk < 4; ++kk) {
        const int c0 = kk * 32 + kb;
        v8h blo = *(const v8h*)(xr + c0);
        v8h bhi = *(const v8h*)(xr + c0 + 16);
        v16h bm = __builtin_shufflevector(blo, bhi, 0,1,2,3,4,5,6,7,8,9,10,11,12,13,14,15);
        accq = __builtin_amdgcn_wmma_f32_16x16x32_f16(false, aq[kk], false, bm,
                                                      (short)0, accq, false, false);
        acck = __builtin_amdgcn_wmma_f32_16x16x32_f16(false, ak[kk], false, bm,
                                                      (short)0, acck, false, false);
      }
      const int n = nt * 16 + lm;
      v8h hq, hk;
#pragma unroll
      for (int r = 0; r < 8; ++r) {
        hq[r] = (_Float16)(accq[r] + bq8[r]);
        hk[r] = (_Float16)(acck[r] + bk8[r]);
      }
      *(v8h*)(sq  + n * QSTR + mcol) = hq;
      *(v8h*)(skf + n * QSTR + mcol) = hk;
    }
  }
  __syncthreads();   // all waves done reading xB and writing sq/skf

  // ---- Phase 2a: rebuild fp32 value row over the xB region ----
  for (int c = 0; c < C_; ++c) {
    const size_t g = ((size_t)b * C_ + c) * HW_ + (size_t)h * W_ + tid;
    sxv[tid * VSTR + c] = feat[g] + pos[g];
  }
  __syncthreads();

  // ---- Phase 2b: scores, softmax, weighted sum ----
  const int w = tid;
  float acc[KWIN];
#pragma unroll
  for (int d = 0; d < KWIN; ++d) acc[d] = 0.0f;

  for (int c0 = 0; c0 < C_; c0 += 8) {
    const v8h qv = *(const v8h*)(sq + w * QSTR + c0);
    float qf[8];
#pragma unroll
    for (int j = 0; j < 8; ++j) qf[j] = (float)qv[j];
#pragma unroll
    for (int d = 0; d < KWIN; ++d) {
      const int wn = w + d - RAD;
      if (wn >= 0 && wn < W_) {
        const v8h kv = *(const v8h*)(skf + wn * QSTR + c0);
        float s = 0.0f;
#pragma unroll
        for (int j = 0; j < 8; ++j) s += qf[j] * (float)kv[j];
        acc[d] += s;
      }
    }
  }

  float mx = acc[0];
#pragma unroll
  for (int d = 1; d < KWIN; ++d) mx = fmaxf(mx, acc[d]);
  float attn[KWIN];
  float sum = 0.0f;
#pragma unroll
  for (int d = 0; d < KWIN; ++d) {
    const float e = __expf(acc[d] - mx);
    attn[d] = e;
    sum += e;
  }
  const float inv = 1.0f / sum;
#pragma unroll
  for (int d = 0; d < KWIN; ++d) attn[d] *= inv;

  float* ob = out + (size_t)b * C_ * HW_ + (size_t)h * W_ + w;
  for (int c0 = 0; c0 < C_; c0 += 4) {
    float o0 = 0.0f, o1 = 0.0f, o2 = 0.0f, o3 = 0.0f;
#pragma unroll
    for (int d = 0; d < KWIN; ++d) {
      const int wn = w + d - RAD;
      if (wn >= 0 && wn < W_) {
        const float a = attn[d];
        const float* vr = sxv + wn * VSTR + c0;
        o0 += a * vr[0];
        o1 += a * vr[1];
        o2 += a * vr[2];
        o3 += a * vr[3];
      }
    }
    ob[(size_t)(c0 + 0) * HW_] = o0;
    ob[(size_t)(c0 + 1) * HW_] = o1;
    ob[(size_t)(c0 + 2) * HW_] = o2;
    ob[(size_t)(c0 + 3) * HW_] = o3;
  }
}

// ---------------------------------------------------------------------------
// Launcher. Workspace (~8.5 MiB): xh f16[N,128] @0, wqh @8388608,
// wkh @8421376, bqs @8454144.
// ---------------------------------------------------------------------------
extern "C" void kernel_launch(void* const* d_in, const int* in_sizes, int n_in,
                              void* d_out, int out_size, void* d_ws, size_t ws_size,
                              hipStream_t stream) {
  const float* feat = (const float*)d_in[0];
  const float* pos  = (const float*)d_in[1];
  const float* Wq   = (const float*)d_in[2];
  const float* bq   = (const float*)d_in[3];
  const float* Wk   = (const float*)d_in[4];
  const float* bk   = (const float*)d_in[5];
  float* out = (float*)d_out;

  char* ws = (char*)d_ws;
  _Float16* xh  = (_Float16*)(ws);
  _Float16* wqh = (_Float16*)(ws + 8388608);
  _Float16* wkh = (_Float16*)(ws + 8421376);
  float*    bqs = (float*)(ws + 8454144);

  prep_w<<<64, 256, 0, stream>>>(Wq, Wk, bq, wqh, wkh, bqs);
  pack_x<<<N_ / 256, 256, 0, stream>>>(feat, pos, xh);

  const size_t shmem = 2 * (size_t)W_ * QSTR * sizeof(_Float16)   // sq + skf
                     + (size_t)W_ * VSTR * sizeof(float);         // xB / sxv union
  // 69632*2 + 132096 = 271360 B  (< 320 KB WGP LDS)
  (void)hipFuncSetAttribute((const void*)attn_fused,
                            hipFuncAttributeMaxDynamicSharedMemorySize, (int)shmem);
  attn_fused<<<B_ * H_, W_, shmem, stream>>>(xh, wqh, wkh, bqs, bk, feat, pos, out);
}
